// SpanModel_53034256171022
// MI455X (gfx1250) — matrix operand order repaired and checked by reading, not compiled, not run on hardware
//
#include <hip/hip_runtime.h>
#include <math.h>

// ---------------- types ----------------
typedef __attribute__((ext_vector_type(16))) __bf16 bf16x16;
typedef __attribute__((ext_vector_type(8)))  __bf16 bf16x8;
typedef __attribute__((ext_vector_type(8)))  float  f32x8;
typedef __attribute__((ext_vector_type(4)))  float  f32x4;
typedef __attribute__((ext_vector_type(2)))  float  f32x2;

#if __has_builtin(__builtin_amdgcn_wmma_f32_16x16x4_f32)
#define HAVE_WMMA_F32X4 1
#else
#define HAVE_WMMA_F32X4 0
#endif

// Problem constants (reference: B=4,S=256,H=256,NE=50000,D=256,Wmax=8,k=8)
#define BB 4
#define SS 256
#define HH 256
#define NE 50000
#define WMAX 8
#define NROWS 8192              // WMAX*BB*SS
#define ENT_ELEMS (NE*HH)       // 12,800,000

struct StarA { float a[64]; };  // normalized star adjacency, 8x8

// ---------------- tiny prep kernels ----------------
__global__ void k_cls(const float* __restrict__ emb, float* __restrict__ out) {
  int t = blockIdx.x * blockDim.x + threadIdx.x;      // 1024
  int b = t >> 8, h = t & 255;
  out[t] = emb[(size_t)b * SS * HH + h];              // emb[b][0][h]
}

__global__ void k_spanmax(const float* __restrict__ emb, float* __restrict__ smax) {
  int t = blockIdx.x * blockDim.x + threadIdx.x;      // B*S*H = 262144
  int s = (t >> 8) & 255;
  const float* base = emb + t;                        // emb[b][s][h] (t = b*65536+s*256+h)
  float cur = base[0];
  smax[t] = cur;                                      // w=0
  #pragma unroll
  for (int w = 1; w < WMAX; ++w) {
    float v = (s + w < SS) ? base[w * HH] : -INFINITY;
    cur = fmaxf(cur, v);
    smax[(size_t)w * (BB*SS*HH) + t] = cur;
  }
}

// bias_wb[w][b][h] = cls[b] @ span_W[256:512] + (w+1)*span_W[512] + span_b
__global__ void k_bias(const float* __restrict__ emb, const float* __restrict__ spanW,
                       const float* __restrict__ spanb, float* __restrict__ bias) {
  int wb = blockIdx.x;          // 0..31
  int w = wb >> 2, b = wb & 3;
  int h = threadIdx.x;          // 0..255
  const float* cls = emb + (size_t)b * SS * HH;
  float acc = spanb[h] + (float)(w + 1) * spanW[512 * HH + h];
  for (int j = 0; j < HH; ++j) acc += cls[j] * spanW[(256 + j) * HH + h];
  bias[wb * HH + h] = acc;
}

__global__ void k_f2b(const float* __restrict__ src, unsigned short* __restrict__ dst, int n) {
  int t = blockIdx.x * blockDim.x + threadIdx.x;
  if (t < n) ((__bf16*)dst)[t] = (__bf16)src[t];
}

// ---------------- span_repr GEMM: (8192x256) @ (256x256) + bias, f32 WMMA ----------------
__global__ void k_span_gemm(const float* __restrict__ A, const float* __restrict__ Wm,
                            const float* __restrict__ bias, float* __restrict__ outRepr,
                            unsigned short* __restrict__ reprB) {
  int lane = threadIdx.x & 31;
  int wave = threadIdx.x >> 5;
  int m0 = blockIdx.x * 16;                       // row tile
  int col = blockIdx.y * 64 + wave * 16 + (lane & 15);
  int kh = lane >> 4;
  int rbase = kh * 8;
  f32x8 acc = {};
#if HAVE_WMMA_F32X4
  const float* arow = A + (size_t)(m0 + (lane & 15)) * HH;
  #pragma unroll 4
  for (int c = 0; c < 64; ++c) {
    int k = c * 4 + kh * 2;
    f32x2 af = *(const f32x2*)(arow + k);                     // A[row][k..k+1]
    f32x2 bf = { Wm[k * HH + col], Wm[(k + 1) * HH + col] };  // B[k..k+1][col]
    acc = __builtin_amdgcn_wmma_f32_16x16x4_f32(false, af, false, bf,
                                                (short)0, acc, false, false);
  }
#else
  for (int k = 0; k < HH; ++k) {
    float bv = Wm[k * HH + col];
    #pragma unroll
    for (int v = 0; v < 8; ++v)
      acc[v] += A[(size_t)(m0 + rbase + v) * HH + k] * bv;
  }
#endif
  int wb = ((m0 >> 10) << 2) | ((m0 >> 8) & 3);   // (w,b) shared within the 16-row tile
  float bv = bias[wb * HH + col];
  #pragma unroll
  for (int v = 0; v < 8; ++v) {
    float val = acc[v] + bv;
    int r = m0 + rbase + v;
    outRepr[(size_t)r * HH + col] = val;
    ((__bf16*)reprB)[(size_t)r * HH + col] = (__bf16)val;
  }
}

// ---------------- fused sims (bf16 WMMA) + streaming top-8 ----------------
// One wave per 16-row tile; 32 entities per iteration (two 16x16 C tiles),
// A fragments register-resident for the whole K=256.
#define SIMS_PAD 36   // row stride in floats: distinct banks for 16 rows, 16B aligned
__global__ void k_sims_topk(const unsigned short* __restrict__ reprB16,
                            const unsigned short* __restrict__ entB16,
                            int* __restrict__ topk) {
  __shared__ float simsT[4][16][SIMS_PAD];
  __shared__ float mVal[4][32][8];
  __shared__ int   mIdx[4][32][8];
  const __bf16* reprB = (const __bf16*)reprB16;
  const __bf16* entB  = (const __bf16*)entB16;
  int lane = threadIdx.x & 31;
  int wv = threadIdx.x >> 5;
  int rt = blockIdx.x * 4 + wv;                 // 0..511 row tile
  int m0 = rt * 16;
  int kh = lane >> 4;
  int r = lane & 15;

  // A fragments for all K=256 (8 chunks of 16x32 bf16), ISA lane layout
  bf16x16 afrag[8];
  const __bf16* abase = reprB + (size_t)(m0 + r) * HH;
  #pragma unroll
  for (int c = 0; c < 8; ++c) {
    bf16x8 lo = *(const bf16x8*)(abase + c * 32 + kh * 8);
    bf16x8 hi = *(const bf16x8*)(abase + c * 32 + 16 + kh * 8);
    #pragma unroll
    for (int j = 0; j < 8; ++j) { afrag[c][j] = lo[j]; afrag[c][j + 8] = hi[j]; }
  }

  float vals[8]; int idxs[8];
  #pragma unroll
  for (int j = 0; j < 8; ++j) { vals[j] = -INFINITY; idxs[j] = 0; }

  const int NT = (NE + 31) / 32;                // 1563 (last tile guarded by id<NE)
  for (int et = 0; et < NT; ++et) {
    int e0 = et * 32;
    const __bf16* eb0 = entB + (size_t)(e0 + r) * HH + kh * 8;
    const __bf16* eb1 = eb0 + 16 * HH;
    // prefetch next iteration's rows (speculative, L2-resident table)
    __builtin_prefetch((const void*)(eb0 + 32 * HH), 0, 1);
    f32x8 acc0 = {}, acc1 = {};
    #pragma unroll
    for (int c = 0; c < 8; ++c) {
      bf16x8 lo0 = *(const bf16x8*)(eb0 + c * 32);
      bf16x8 hi0 = *(const bf16x8*)(eb0 + c * 32 + 16);
      bf16x8 lo1 = *(const bf16x8*)(eb1 + c * 32);
      bf16x8 hi1 = *(const bf16x8*)(eb1 + c * 32 + 16);
      bf16x16 bf0, bf1;
      #pragma unroll
      for (int j = 0; j < 8; ++j) {
        bf0[j] = lo0[j]; bf0[j + 8] = hi0[j];
        bf1[j] = lo1[j]; bf1[j + 8] = hi1[j];
      }
      acc0 = __builtin_amdgcn_wmma_f32_16x16x32_bf16(false, afrag[c], false, bf0,
                                                     (short)0, acc0, false, false);
      acc1 = __builtin_amdgcn_wmma_f32_16x16x32_bf16(false, afrag[c], false, bf1,
                                                     (short)0, acc1, false, false);
    }
    // spill both C tiles (lane: col = lane&15, rows v + 8*kh)
    #pragma unroll
    for (int v = 0; v < 8; ++v) {
      simsT[wv][v + 8 * kh][r]      = acc0[v];
      simsT[wv][v + 8 * kh][16 + r] = acc1[v];
    }
    asm volatile("s_wait_dscnt 0" ::: "memory");
    // each row handled by 2 lanes; lane scans 16 columns (vectorized DS reads)
    int cbase = kh * 16;
    const f32x4* rowp = (const f32x4*)&simsT[wv][r][cbase];
    f32x4 q[4];
    #pragma unroll
    for (int qq = 0; qq < 4; ++qq) q[qq] = rowp[qq];
    #pragma unroll
    for (int c = 0; c < 16; ++c) {
      float v = q[c >> 2][c & 3];
      int id = e0 + cbase + c;
      if (v > vals[7] && id < NE) {
        float nv = v; int ni = id;
        #pragma unroll
        for (int j = 0; j < 8; ++j) {
          if (nv > vals[j]) {
            float tv = vals[j]; int ti = idxs[j];
            vals[j] = nv; idxs[j] = ni; nv = tv; ni = ti;
          }
        }
      }
    }
  }
  // merge the two partial lists per row
  #pragma unroll
  for (int j = 0; j < 8; ++j) { mVal[wv][lane][j] = vals[j]; mIdx[wv][lane][j] = idxs[j]; }
  asm volatile("s_wait_dscnt 0" ::: "memory");
  if (lane < 16) {
    float fv[8]; int fi[8];
    #pragma unroll
    for (int j = 0; j < 8; ++j) { fv[j] = mVal[wv][lane][j]; fi[j] = mIdx[wv][lane][j]; }
    #pragma unroll
    for (int j = 0; j < 8; ++j) {
      float nv = mVal[wv][lane + 16][j]; int ni = mIdx[wv][lane + 16][j];
      if (nv > fv[7]) {
        #pragma unroll
        for (int p = 0; p < 8; ++p) {
          if (nv > fv[p]) {
            float tv = fv[p]; int ti = fi[p];
            fv[p] = nv; fi[p] = ni; nv = tv; ni = ti;
          }
        }
      }
    }
    int row = m0 + lane;
    #pragma unroll
    for (int j = 0; j < 8; ++j) topk[row * 8 + j] = fi[j];
  }
}

// ---------------- fused 2-layer star-GCN (2 spans per workgroup) ----------------
__global__ void k_gcn(const int* __restrict__ topk, const float* __restrict__ ent,
                      const float* __restrict__ W1, const float* __restrict__ b1v,
                      const float* __restrict__ W2, const float* __restrict__ b2v,
                      float* __restrict__ out, StarA An) {
  __shared__ float X[16][HH];
  __shared__ float Y[16][HH];
  int pair = blockIdx.x;                // 0..4095 (two spans)
  int t = threadIdx.x;                  // 128
  for (int i = t; i < 16 * HH; i += 128) {
    int rr = i >> 8, c = i & 255;
    int n = pair * 2 + (rr >> 3);
    int e = topk[n * 8 + (rr & 7)];
    X[rr][c] = ent[(size_t)e * HH + c];
  }
  __syncthreads();
  int lane = t & 31, wv = t >> 5;
  int kh = lane >> 4;
  int rsel = lane & 15;
  // ---- layer 1 GEMM: Y = X @ W1 ----
  for (int nt = 0; nt < 4; ++nt) {
    int col = wv * 64 + nt * 16 + rsel;
    f32x8 acc = {};
#if HAVE_WMMA_F32X4
    #pragma unroll 4
    for (int c = 0; c < 64; ++c) {
      int k = c * 4 + kh * 2;
      f32x2 af = { X[rsel][k], X[rsel][k + 1] };
      f32x2 bf = { W1[k * HH + col], W1[(k + 1) * HH + col] };
      acc = __builtin_amdgcn_wmma_f32_16x16x4_f32(false, af, false, bf,
                                                  (short)0, acc, false, false);
    }
#else
    for (int k = 0; k < HH; ++k) {
      float bv = W1[k * HH + col];
      #pragma unroll
      for (int v = 0; v < 8; ++v) acc[v] += X[kh * 8 + v][k] * bv;
    }
#endif
    #pragma unroll
    for (int v = 0; v < 8; ++v) Y[v + 8 * kh][col] = acc[v];
  }
  __syncthreads();
  // ---- star mix + bias + relu (into X) ----
  for (int i = t; i < 16 * HH; i += 128) {
    int rr = i >> 8, c = i & 255;
    int blk = rr & 8, ri = rr & 7;
    float s = b1v[c];
    #pragma unroll
    for (int j = 0; j < 8; ++j) s += An.a[ri * 8 + j] * Y[blk + j][c];
    X[rr][c] = fmaxf(s, 0.0f);
  }
  __syncthreads();
  // ---- layer 2 GEMM: Y = h @ W2 ----
  for (int nt = 0; nt < 4; ++nt) {
    int col = wv * 64 + nt * 16 + rsel;
    f32x8 acc = {};
#if HAVE_WMMA_F32X4
    #pragma unroll 4
    for (int c = 0; c < 64; ++c) {
      int k = c * 4 + kh * 2;
      f32x2 af = { X[rsel][k], X[rsel][k + 1] };
      f32x2 bf = { W2[k * HH + col], W2[(k + 1) * HH + col] };
      acc = __builtin_amdgcn_wmma_f32_16x16x4_f32(false, af, false, bf,
                                                  (short)0, acc, false, false);
    }
#else
    for (int k = 0; k < HH; ++k) {
      float bv = W2[k * HH + col];
      #pragma unroll
      for (int v = 0; v < 8; ++v) acc[v] += X[kh * 8 + v][k] * bv;
    }
#endif
    #pragma unroll
    for (int v = 0; v < 8; ++v) Y[v + 8 * kh][col] = acc[v];
  }
  __syncthreads();
  // ---- star mix + bias -> output ----
  for (int i = t; i < 16 * HH; i += 128) {
    int rr = i >> 8, c = i & 255;
    int blk = rr & 8, ri = rr & 7;
    float s = b2v[c];
    #pragma unroll
    for (int j = 0; j < 8; ++j) s += An.a[ri * 8 + j] * Y[blk + j][c];
    int n = pair * 2 + (rr >> 3);
    out[((size_t)n * 8 + (rr & 7)) * HH + c] = s;
  }
}

// ---------------- host launch ----------------
extern "C" void kernel_launch(void* const* d_in, const int* in_sizes, int n_in,
                              void* d_out, int out_size, void* d_ws, size_t ws_size,
                              hipStream_t stream) {
  (void)in_sizes; (void)n_in; (void)out_size; (void)ws_size;
  const float* emb   = (const float*)d_in[0];
  const float* ent   = (const float*)d_in[1];
  const float* spanW = (const float*)d_in[2];
  const float* spanb = (const float*)d_in[3];
  const float* W1    = (const float*)d_in[4];
  const float* b1    = (const float*)d_in[5];
  const float* W2    = (const float*)d_in[6];
  const float* b2    = (const float*)d_in[7];
  // d_in[8] = k (constant 8)

  char* ws = (char*)d_ws;
  float*          spanMax = (float*)(ws + 0);                      // 8 MB
  float*          biasWB  = (float*)(ws + 8388608);                // 32 KB
  int*            topk    = (int*)  (ws + 8421376);                // 256 KB
  unsigned short* entB    = (unsigned short*)(ws + 8683520);       // 25.6 MB
  unsigned short* reprB   = (unsigned short*)(ws + 34283520);      // 4 MB

  float* out     = (float*)d_out;
  float* outCls  = out;                                  // 1024
  float* outRepr = out + 1024;                           // 8192*256
  float* outSub  = out + 1024 + (size_t)NROWS * HH;      // 8192*8*256

  k_cls    <<<4,    256, 0, stream>>>(emb, outCls);
  k_spanmax<<<1024, 256, 0, stream>>>(emb, spanMax);
  k_bias   <<<32,   256, 0, stream>>>(emb, spanW, spanb, biasWB);
  k_f2b    <<<ENT_ELEMS / 256, 256, 0, stream>>>(ent, entB, ENT_ELEMS);
  k_span_gemm<<<dim3(NROWS / 16, 4), 128, 0, stream>>>(spanMax, spanW, biasWB, outRepr, reprB);
  k_sims_topk<<<NROWS / 64, 128, 0, stream>>>(reprB, entB, topk);

  StarA An;
  {
    float A[8][8] = {};
    for (int j = 1; j < 8; ++j) { A[0][j] = 1.0f; A[j][0] = 1.0f; }
    for (int i = 0; i < 8; ++i) A[i][i] += 1.0f;
    float deg[8];
    for (int i = 0; i < 8; ++i) { float s = 0; for (int j = 0; j < 8; ++j) s += A[i][j]; deg[i] = s; }
    for (int i = 0; i < 8; ++i)
      for (int j = 0; j < 8; ++j)
        An.a[i * 8 + j] = A[i][j] / sqrtf(deg[i] * deg[j]);
  }
  k_gcn<<<NROWS / 2, 128, 0, stream>>>(topk, ent, W1, b1, W2, b2, outSub, An);
}
